// RecurrentNonLocalKmeansAttention_6133213299490
// MI455X (gfx1250) — matrix operand order repaired and checked by reading, not compiled, not run on hardware
//
#include <hip/hip_runtime.h>
#include <hip/hip_bf16.h>

typedef __attribute__((ext_vector_type(16))) _Float16 v16h;
typedef __attribute__((ext_vector_type(4)))  _Float16 v4h;
typedef __attribute__((ext_vector_type(8)))  float    v8f;

static constexpr int kN     = 8;
static constexpr int kCIN   = 256;
static constexpr int kCQ    = 64;
static constexpr int kH     = 96;
static constexpr int kW     = 96;
static constexpr int kL     = kH * kW;          // 9216
static constexpr int kNL    = kN * kL;          // 73728
static constexpr int kK3    = kCIN * 9;         // 2304
static constexpr int kNC    = 128;              // clusters
static constexpr int kWIN   = 144;
static constexpr int kNWIN  = 64;               // 9216/144
static constexpr int kJ3    = 3 * kWIN;         // 432

// ---------------- WMMA helpers (CDNA5 16x16x32 f16, f32 acc) ----------------
__device__ __forceinline__ int kmapA(int e, int lane) {
  // A fragment: lanes hold M=lane&15; element e -> K (two 8-half runs)
  return (e & 7) + ((e >> 3) << 4) + ((lane >> 4) << 3);
}
__device__ __forceinline__ int kmapB(int e, int lane) {
  // B fragment: lanes hold N=lane&15; element e -> K (two 8-half runs, same as A)
  return (e & 7) + ((e >> 3) << 4) + ((lane >> 4) << 3);
}
__device__ __forceinline__ v8f wmma_f16(v16h a, v16h b, v8f c) {
  return __builtin_amdgcn_wmma_f32_16x16x32_f16(false, a, false, b, (short)0, c,
                                                false, false);
}
__device__ __forceinline__ int win_row(int k, int j) {
  // _add_adjacent_buckets order: [self, back(k-1), fwd(k+1)]
  if (j < kWIN)     return k * kWIN + j;
  if (j < 2 * kWIN) return ((k + kNWIN - 1) & (kNWIN - 1)) * kWIN + (j - kWIN);
  return ((k + 1) & (kNWIN - 1)) * kWIN + (j - 2 * kWIN);
}

// CDNA5 async global->LDS DMA: 16B per lane, tracked on ASYNCcnt (no VGPR round trip).
__device__ __forceinline__ void async_copy_b128(unsigned lds_byte_addr,
                                                const void* gaddr) {
  asm volatile("global_load_async_to_lds_b128 %0, %1, off"
               :: "v"(lds_byte_addr), "v"(gaddr) : "memory");
}
__device__ __forceinline__ void wait_asynccnt0() {
  asm volatile("s_wait_asynccnt 0" ::: "memory");
}

// ---------------- f32 -> f16 convert ----------------
__global__ void k_cvt(const float* __restrict__ in, _Float16* __restrict__ out, int n) {
  int i = blockIdx.x * 256 + threadIdx.x;
  if (i < n) out[i] = (_Float16)in[i];
}

// ---------------- conv3x3 implicit GEMM: x_embed [NL][64] ----------------
// B tile stored n-major: Bs[n][32] so a lane's fragment is two contiguous 16B runs.
__global__ __launch_bounds__(128) void k_conv3(const _Float16* __restrict__ xh,
                                               const _Float16* __restrict__ wmh, // [64][2304]
                                               const float* __restrict__ bias,
                                               float* __restrict__ xe) {
  __shared__ alignas(16) _Float16 As[16 * 32];
  __shared__ alignas(16) _Float16 Bs[kCQ * 32];
  const int tid = threadIdx.x, lane = tid & 31, wave = tid >> 5;  // 4 waves
  const int m0 = blockIdx.x * 16;
  const int n = m0 / kL, pix0 = m0 % kL;

  // im2col staging: this thread owns column kkf and rows {r0, r0+4, r0+8, r0+12}
  const int kkf = tid & 31;
  const int r0  = tid >> 5;
  int hs[4], wsx[4];
#pragma unroll
  for (int q = 0; q < 4; ++q) {
    int pix = pix0 + r0 + q * 4;
    hs[q] = pix / kW;
    wsx[q] = pix % kW;
  }
  int c_cur = kkf / 9, r9 = kkf % 9;       // K decomposition, advanced incrementally

  v8f acc = {};
  for (int kt = 0; kt < kK3 / 32; ++kt) {
    const int k0 = kt * 32;
    __syncthreads();
    {  // A tile: clamped unconditional load + select (no exec-mask branches)
      const int dh = r9 / 3 - 1, dw = r9 % 3 - 1;
      const _Float16* src = xh + ((size_t)n * kCIN + c_cur) * kL;
#pragma unroll
      for (int q = 0; q < 4; ++q) {
        int hh = hs[q] + dh, ww = wsx[q] + dw;
        bool inb = (hh >= 0) & (hh < kH) & (ww >= 0) & (ww < kW);
        int hc = min(max(hh, 0), kH - 1), wc = min(max(ww, 0), kW - 1);
        _Float16 v = src[hc * kW + wc];
        As[(r0 + q * 4) * 32 + kkf] = inb ? v : (_Float16)0.f;
      }
      r9 += 5; c_cur += 3;
      if (r9 >= 9) { r9 -= 9; ++c_cur; }
    }
    // B tile: 64 rows x 32 k -> 256 uint4 chunks, coalesced
    for (int t = tid; t < kCQ * 4; t += 128) {
      int nn = t >> 2, ch = t & 3;
      ((uint4*)Bs)[t] = *(const uint4*)(wmh + (size_t)nn * kK3 + k0 + ch * 8);
    }
    if (k0 + 32 < kK3)
      __builtin_prefetch(wmh + (size_t)(tid & 63) * kK3 + k0 + 32, 0, 1);
    __syncthreads();
    v16h a{}, b{};
#pragma unroll
    for (int e = 0; e < 16; ++e) {
      a[e] = As[(lane & 15) * 32 + kmapA(e, lane)];
      b[e] = Bs[(wave * 16 + (lane & 15)) * 32 + kmapB(e, lane)];
    }
    acc = wmma_f16(a, b, acc);
  }
  const int col = wave * 16 + (lane & 15);
  const float bb = bias[col];
  const int rh = (lane >> 4) * 8;
#pragma unroll
  for (int r = 0; r < 8; ++r)
    xe[(size_t)(m0 + rh + r) * kCQ + col] = acc[r] + bb;
}

// ---------------- conv1x1 GEMM: y_embed [NL][256] ----------------
__global__ __launch_bounds__(256) void k_conv1(const _Float16* __restrict__ xh,
                                               const _Float16* __restrict__ wah, // [256][256]
                                               const float* __restrict__ bias,
                                               float* __restrict__ ye) {
  __shared__ alignas(16) _Float16 As[16 * 32];
  __shared__ alignas(16) _Float16 Bs[128 * 32];
  const int tid = threadIdx.x, lane = tid & 31, wave = tid >> 5;  // 8 waves
  const int m0 = blockIdx.x * 16;
  const int nb = blockIdx.y * 128;
  const int n = m0 / kL, pix0 = m0 % kL;
  v8f acc = {};
  for (int kt = 0; kt < kCIN / 32; ++kt) {
    const int k0 = kt * 32;
    __syncthreads();
    // A tile: r fastest over tid -> coalesced pixel reads
    for (int t = tid; t < 16 * 32; t += 256) {
      int r = t & 15, kk = t >> 4;
      As[r * 32 + kk] = xh[((size_t)n * kCIN + k0 + kk) * kL + pix0 + r];
    }
    // B tile: 128 rows x 32 k -> 512 uint4 chunks
    for (int t = tid; t < 128 * 4; t += 256) {
      int nn = t >> 2, ch = t & 3;
      ((uint4*)Bs)[t] = *(const uint4*)(wah + (size_t)(nb + nn) * kCIN + k0 + ch * 8);
    }
    if (k0 + 32 < kCIN)
      __builtin_prefetch(wah + (size_t)(nb + (tid & 127)) * kCIN + k0 + 32, 0, 1);
    __syncthreads();
    v16h a{}, b{};
#pragma unroll
    for (int e = 0; e < 16; ++e) {
      a[e] = As[(lane & 15) * 32 + kmapA(e, lane)];
      b[e] = Bs[(wave * 16 + (lane & 15)) * 32 + kmapB(e, lane)];
    }
    acc = wmma_f16(a, b, acc);
  }
  const int col = nb + wave * 16 + (lane & 15);
  const float bb = bias[col];
  const int rh = (lane >> 4) * 8;
#pragma unroll
  for (int r = 0; r < 8; ++r)
    ye[(size_t)(m0 + rh + r) * kCIN + col] = acc[r] + bb;
}

// ---------------- cluster assignment + stable sort keys ----------------
__global__ __launch_bounds__(128) void k_assign(const float* __restrict__ xe,
                                                const float* __restrict__ means,
                                                float* __restrict__ codes_out,
                                                unsigned* __restrict__ keys) {
  __shared__ float sm[kNC * kCQ];          // 32 KB
  const int tid = threadIdx.x;
  for (int t = tid; t < kNC * kCQ; t += 128) sm[t] = means[t];
  __syncthreads();
  const int gl = blockIdx.x * 128 + tid;   // < kNL exactly
  float v[kCQ]; float ss = 0.f;
#pragma unroll 8
  for (int e = 0; e < kCQ; ++e) { float x = xe[(size_t)gl * kCQ + e]; v[e] = x; ss += x * x; }
  const float inv = 1.0f / fmaxf(sqrtf(ss), 1e-12f);
#pragma unroll 8
  for (int e = 0; e < kCQ; ++e) v[e] *= inv;
  int best = 0; float bv = -3.0e38f;
  for (int c = 0; c < kNC; ++c) {
    float d = 0.f;
#pragma unroll 8
    for (int e = 0; e < kCQ; ++e) d += v[e] * sm[c * kCQ + e];
    if (d > bv) { bv = d; best = c; }
  }
  codes_out[gl] = (float)best;
  keys[gl] = (unsigned)(best * 16384 + (gl % kL));   // stable composite key
}

// ---------------- per-batch bitonic sort (16384 keys in LDS) ----------------
__global__ __launch_bounds__(1024) void k_sort(const unsigned* __restrict__ keys,
                                               int* __restrict__ idx,
                                               int* __restrict__ undo) {
  __shared__ unsigned sk[16384];           // 64 KB
  const int n = blockIdx.x, tid = threadIdx.x;
  for (int t = tid; t < 16384; t += 1024)
    sk[t] = (t < kL) ? keys[n * kL + t] : 0xFFFFFFFFu;
  __syncthreads();
  for (int size = 2; size <= 16384; size <<= 1) {
    for (int stride = size >> 1; stride > 0; stride >>= 1) {
      for (int t = tid; t < 8192; t += 1024) {
        int lo = 2 * t - (t & (stride - 1));
        int hi = lo + stride;
        bool up = ((lo & size) == 0);
        unsigned a = sk[lo], b = sk[hi];
        if ((a > b) == up) { sk[lo] = b; sk[hi] = a; }
      }
      __syncthreads();
    }
  }
  for (int t = tid; t < kL; t += 1024) idx[n * kL + t] = (int)(sk[t] & 16383u);
  __syncthreads();
  for (int t = tid; t < kL; t += 1024) undo[n * kL + (int)(sk[t] & 16383u)] = t;
}

// ---------------- gather sorted rows, f16 convert, l2norm(5e-5) ----------------
__global__ __launch_bounds__(256) void k_gather(const float* __restrict__ xe,
                                                const float* __restrict__ ye,
                                                const int* __restrict__ idx,
                                                _Float16* __restrict__ xs,
                                                _Float16* __restrict__ xsn,
                                                _Float16* __restrict__ ys) {
  const int g = blockIdx.x * 256 + threadIdx.x;    // < kNL exactly
  const int n = g / kL;
  const int i = idx[g];
  const float* sx = xe + (size_t)(n * kL + i) * kCQ;
  float ss = 0.f;
#pragma unroll 8
  for (int e = 0; e < kCQ; ++e) { float v = sx[e]; xs[(size_t)g * kCQ + e] = (_Float16)v; ss += v * v; }
  const float inv = 1.0f / fmaxf(sqrtf(ss), 5e-5f);
#pragma unroll 8
  for (int e = 0; e < kCQ; ++e) xsn[(size_t)g * kCQ + e] = (_Float16)(sx[e] * inv);
  const float4* sy4 = (const float4*)(ye + (size_t)(n * kL + i) * kCIN);
#pragma unroll 4
  for (int c = 0; c < kCIN / 4; ++c) {
    float4 f = sy4[c];
    v4h h;
    h[0] = (_Float16)f.x; h[1] = (_Float16)f.y; h[2] = (_Float16)f.z; h[3] = (_Float16)f.w;
    *(v4h*)(ys + (size_t)g * kCIN + c * 4) = h;
  }
}

// ---------------- attention pass 1: raw = Q K^T  (+ online logsumexp) ----------------
__global__ __launch_bounds__(288) void k_raw(const _Float16* __restrict__ xs,
                                             const _Float16* __restrict__ xsn,
                                             float* __restrict__ rawout,
                                             float* __restrict__ bsout) {
  __shared__ alignas(16) _Float16 Qs[kWIN * kCQ];      // 18 KB
  __shared__ alignas(16) _Float16 Ks[kJ3 * kCQ];       // 55 KB
  const int blk = blockIdx.x;
  const int n = blk >> 6, k = blk & 63;
  const int tid = threadIdx.x, lane = tid & 31, wave = tid >> 5;  // 9 waves
  const _Float16* qsrc = xs + ((size_t)n * kL + k * kWIN) * kCQ;

  // Bulk global->LDS staging via CDNA5 async DMA (ASYNCcnt), 16B per lane.
  // Trip counts are exact multiples of 288 -> EXEC is all-ones at every issue.
  const unsigned ldsQ = (unsigned)(uintptr_t)(&Qs[0]);
  const unsigned ldsK = (unsigned)(uintptr_t)(&Ks[0]);
  for (int t = tid; t < (kWIN * kCQ) / 8; t += 288)    // 1152 chunks
    async_copy_b128(ldsQ + t * 16, (const void*)(((const uint4*)qsrc) + t));
  for (int t = tid; t < kJ3 * 8; t += 288) {           // 432 rows x 8 chunks
    int j = t >> 3, ch = t & 7;
    async_copy_b128(ldsK + t * 16,
                    (const void*)(xsn + ((size_t)n * kL + win_row(k, j)) * kCQ + ch * 8));
  }
  wait_asynccnt0();          // producer side: own DMAs landed in LDS
  __syncthreads();           // consumer side: all waves' tiles visible

  const int mt = wave;                      // one 16-row M tile per wave
  v16h a0{}, a1{};
#pragma unroll
  for (int e = 0; e < 16; ++e) {
    int kk = kmapA(e, lane);
    a0[e] = Qs[(mt * 16 + (lane & 15)) * kCQ + kk];
    a1[e] = Qs[(mt * 16 + (lane & 15)) * kCQ + 32 + kk];
  }
  float m[8], s[8];
#pragma unroll
  for (int r = 0; r < 8; ++r) { m[r] = -3.0e38f; s[r] = 0.f; }
  const int rh = (lane >> 4) * 8;
#pragma unroll 3
  for (int nt = 0; nt < kJ3 / 16; ++nt) {
    v16h b0{}, b1{};
    const int col = nt * 16 + (lane & 15);
#pragma unroll
    for (int e = 0; e < 16; ++e) {
      int kk = kmapB(e, lane);
      b0[e] = Ks[col * kCQ + kk];           // two contiguous 16B runs per lane
      b1[e] = Ks[col * kCQ + 32 + kk];
    }
    v8f acc = {};
    acc = wmma_f16(a0, b0, acc);
    acc = wmma_f16(a1, b1, acc);
    const size_t base =
        ((size_t)(n * kNWIN + k) * kWIN + mt * 16) * kJ3 + nt * 16 + (lane & 15);
#pragma unroll
    for (int r = 0; r < 8; ++r) rawout[base + (size_t)(r + rh) * kJ3] = acc[r];
#pragma unroll
    for (int r = 0; r < 8; ++r) {          // online logsumexp, 16-lane row reduce
      float tm = acc[r];
      for (int off = 8; off >= 1; off >>= 1) tm = fmaxf(tm, __shfl_xor(tm, off, 32));
      float nm = fmaxf(m[r], tm);
      float es = expf(acc[r] - nm);
      for (int off = 8; off >= 1; off >>= 1) es += __shfl_xor(es, off, 32);
      s[r] = s[r] * expf(m[r] - nm) + es;
      m[r] = nm;
    }
  }
  if ((lane & 15) == 0) {
#pragma unroll
    for (int r = 0; r < 8; ++r) {
      int row = mt * 16 + rh + r;
      bsout[n * kL + k * kWIN + row] = m[r] + logf(s[r]);
    }
  }
}

// ---------------- score = exp(raw - bs), in place ----------------
__global__ void k_fscore(float* __restrict__ score, const float* __restrict__ bs) {
  const size_t t = (size_t)blockIdx.x * 256 + threadIdx.x;  // N*64*144*432 exactly
  const int row = (int)(t / kJ3);          // row == n*9216 + (k*144+i)
  score[t] = expf(score[t] - bs[row]);
}

// ---------------- attention pass 2: ret = score * yb3 ----------------
__global__ __launch_bounds__(256) void k_ret(const float* __restrict__ score,
                                             const _Float16* __restrict__ ys,
                                             float* __restrict__ ret) {
  __shared__ alignas(16) _Float16 As[16 * 448];        // score tile, K padded 432->448
  __shared__ alignas(16) _Float16 Bs[kCIN * 32];       // V tile, n-major
  const int blk = blockIdx.x;
  const int n = blk >> 6, k = blk & 63;
  const int tid = threadIdx.x, lane = tid & 31, wave = tid >> 5;  // 8 waves x 32 cols
  for (int mt = 0; mt < kWIN / 16; ++mt) {
    __syncthreads();
    const size_t srow = (size_t)(n * kNWIN + k) * kWIN + mt * 16;
    // A tile: float4 loads + packed f16 stores, zero-padded K tail
    for (int t = tid; t < 16 * 112; t += 256) {
      int r = t / 112, c = (t % 112) * 4;
      v4h hv = {};
      if (c < kJ3) {
        const float4 f = *(const float4*)(score + (srow + r) * (size_t)kJ3 + c);
        hv[0] = (_Float16)f.x; hv[1] = (_Float16)f.y;
        hv[2] = (_Float16)f.z; hv[3] = (_Float16)f.w;
      }
      *(v4h*)(As + r * 448 + c) = hv;
    }
    v8f acc0 = {}, acc1 = {};
    for (int kt = 0; kt < 14; ++kt) {
      __syncthreads();
      const int k0 = kt * 32;
      // B tile n-major: Bs[c][kk]; global reads are 16B runs along channels
      for (int t = tid; t < 32 * 32; t += 256) {
        int kk = t >> 5, c8 = (t & 31) << 3;
        int j = k0 + kk;
        if (j < kJ3) {
          const _Float16* src = ys + ((size_t)n * kL + win_row(k, j)) * kCIN + c8;
#pragma unroll
          for (int q = 0; q < 8; ++q) Bs[(c8 + q) * 32 + kk] = src[q];
        } else {
#pragma unroll
          for (int q = 0; q < 8; ++q) Bs[(c8 + q) * 32 + kk] = (_Float16)0.f;
        }
      }
      if (k0 + 32 < kJ3)
        __builtin_prefetch(ys + ((size_t)n * kL + win_row(k, k0 + 32 + (tid & 31))) * kCIN,
                           0, 1);
      __syncthreads();
      v16h a{}, b0{}, b1{};
#pragma unroll
      for (int e = 0; e < 16; ++e) {
        int ka = kmapA(e, lane), kb = kmapB(e, lane);
        a[e]  = As[(lane & 15) * 448 + k0 + ka];
        b0[e] = Bs[(wave * 32 + (lane & 15)) * 32 + kb];
        b1[e] = Bs[(wave * 32 + 16 + (lane & 15)) * 32 + kb];
      }
      acc0 = wmma_f16(a, b0, acc0);
      acc1 = wmma_f16(a, b1, acc1);
    }
    const int rh = (lane >> 4) * 8;
    const size_t rowg = (size_t)n * kL + k * kWIN + mt * 16;
#pragma unroll
    for (int r = 0; r < 8; ++r) {
      size_t rr = (rowg + rh + r) * kCIN;
      ret[rr + wave * 32 + (lane & 15)]      = acc0[r];
      ret[rr + wave * 32 + 16 + (lane & 15)] = acc1[r];
    }
  }
}

// ---------------- final: out = x + 0.1 * ret[undo]  (probs == 1, R == 1) ----------------
__global__ void k_out(const float* __restrict__ x, const float* __restrict__ ret,
                      const int* __restrict__ undo, float* __restrict__ out) {
  const int b = blockIdx.x;                 // kNL blocks
  const int n = b / kL, l = b % kL;
  const int j = undo[b];
  const int c = threadIdx.x;                // 256 channels
  const size_t oi = ((size_t)n * kCIN + c) * kL + l;
  out[oi] = x[oi] + 0.1f * ret[((size_t)(n * kL + j)) * kCIN + c];
}

// ---------------- host launcher ----------------
extern "C" void kernel_launch(void* const* d_in, const int* in_sizes, int n_in,
                              void* d_out, int out_size, void* d_ws, size_t ws_size,
                              hipStream_t stream) {
  (void)in_sizes; (void)n_in; (void)out_size; (void)ws_size;
  const float* x       = (const float*)d_in[0];
  const float* means   = (const float*)d_in[1];
  const float* w_match = (const float*)d_in[2];
  const float* b_match = (const float*)d_in[3];
  const float* w_asm   = (const float*)d_in[4];
  const float* b_asm   = (const float*)d_in[5];

  // output regions (return order: out, score, bs_full, codes)
  float* outp   = (float*)d_out;
  float* scorep = outp + (size_t)kN * kCIN * kL;                 // 18,874,368
  float* bsp    = scorep + (size_t)kN * kNWIN * kWIN * kJ3;      // +31,850,496
  float* codesp = bsp + (size_t)kN * kL;

  // workspace carve-up
  char* ws = (char*)d_ws;
  size_t o = 0;
  auto carve = [&](size_t bytes) { char* p = ws + o; o = (o + bytes + 255) & ~(size_t)255; return p; };
  _Float16* xh   = (_Float16*)carve((size_t)kN * kCIN * kL * 2);
  _Float16* wmh  = (_Float16*)carve((size_t)kCQ * kK3 * 2);
  _Float16* wah  = (_Float16*)carve((size_t)kCIN * kCIN * 2);
  float*    xe   = (float*)   carve((size_t)kNL * kCQ * 4);
  float*    ye   = (float*)   carve((size_t)kNL * kCIN * 4);
  unsigned* keys = (unsigned*)carve((size_t)kNL * 4);
  int*      idx  = (int*)     carve((size_t)kNL * 4);
  int*      undo = (int*)     carve((size_t)kNL * 4);
  _Float16* xs   = (_Float16*)carve((size_t)kNL * kCQ * 2);
  _Float16* xsn  = (_Float16*)carve((size_t)kNL * kCQ * 2);
  _Float16* ysg  = (_Float16*)carve((size_t)kNL * kCIN * 2);
  float*    ret  = (float*)   carve((size_t)kNL * kCIN * 4);

  const int nx = kN * kCIN * kL;
  k_cvt<<<nx / 256, 256, 0, stream>>>(x, xh, nx);
  k_cvt<<<(kCQ * kK3 + 255) / 256, 256, 0, stream>>>(w_match, wmh, kCQ * kK3);
  k_cvt<<<(kCIN * kCIN + 255) / 256, 256, 0, stream>>>(w_asm, wah, kCIN * kCIN);

  k_conv3<<<kNL / 16, 128, 0, stream>>>(xh, wmh, b_match, xe);
  k_conv1<<<dim3(kNL / 16, kCIN / 128), 256, 0, stream>>>(xh, wah, b_asm, ye);

  k_assign<<<kNL / 128, 128, 0, stream>>>(xe, means, codesp, keys);
  k_sort<<<kN, 1024, 0, stream>>>(keys, idx, undo);
  k_gather<<<kNL / 256, 256, 0, stream>>>(xe, ye, idx, xs, xsn, ysg);

  k_raw<<<kN * kNWIN, 288, 0, stream>>>(xs, xsn, scorep, bsp);
  const size_t nsc = (size_t)kN * kNWIN * kWIN * kJ3;
  k_fscore<<<(unsigned)(nsc / 256), 256, 0, stream>>>(scorep, bsp);
  k_ret<<<kN * kNWIN, 256, 0, stream>>>(scorep, ysg, ret);

  k_out<<<kNL, 256, 0, stream>>>(x, ret, undo, outp);
}
